// GAF_317827580564
// MI455X (gfx1250) — matrix operand order repaired and checked by reading, not compiled, not run on hardware
//
#include <hip/hip_runtime.h>
#include <hip/hip_bf16.h>

typedef _Float16 v16h __attribute__((ext_vector_type(16)));
typedef _Float16 v8h  __attribute__((ext_vector_type(8)));
typedef float    v8f  __attribute__((ext_vector_type(8)));

union HV16 { v16h v; v8h h[2]; };

#define KB   32
#define KC   256
#define KP   1024
#define KOUT 512
#define KM   (KB*KP)            // 32768 rows (B*P)
#define BN_INV 0.99999500003749937f  // 1/sqrt(1+1e-5)

typedef unsigned long long ull;

__device__ __forceinline__ float seluf(float x){
  const float sc = 1.0507009873554805f;
  const float al = 1.6732632423543772f;
  return x > 0.f ? sc * x : sc * al * (__expf(x) - 1.f);
}

// ---------------------------------------------------------------------------
// WMMA GEMM:  C[M,N] = epi( A[M,K] @ B[N,K]^T )  both operands K-contiguous.
// block = 128 threads (4 waves), block tile 128x128, wave tile 64x64 (4x4 WMMA)
// EPI: 0 = acc (+bias)            -> f16
//      1 = selu(acc+bias)         -> f16
//      2 = alpha*acc              -> f16
//      3 = relu(g[n]*INV*(gate[m]*acc+bias)+beta[n])        -> f16
//      4 = relu(g[n]*INV*(acc+bias)+beta[n]) -> f32 stored transposed [b,n,p]
// ---------------------------------------------------------------------------
template<int EPI>
__global__ __launch_bounds__(128)
void gemm_wmma(const _Float16* __restrict__ A, ull sA, int lda,
               const _Float16* __restrict__ Bw, ull sB, int ldb,
               _Float16* __restrict__ Co, ull sC, int ldc,
               float* __restrict__ Cf, int K,
               const float* __restrict__ bias,
               const float* __restrict__ gamma,
               const float* __restrict__ beta,
               const float* __restrict__ gate,
               float alpha)
{
  const int wave = threadIdx.x >> 5;
  const int lane = threadIdx.x & 31;
  const int lh   = lane & 15;
  const int hi   = lane >> 4;          // half-wave select
  const int z    = blockIdx.z;
  const int bm   = blockIdx.x * 128 + (wave >> 1) * 64;
  const int bn   = blockIdx.y * 128 + (wave & 1) * 64;

  const _Float16* Ab = A  + (ull)z * sA;
  const _Float16* Bb = Bw + (ull)z * sB;

  const _Float16* Ar[4];
  const _Float16* Br[4];
#pragma unroll
  for (int i = 0; i < 4; i++) Ar[i] = Ab + (size_t)(bm + i*16 + lh) * lda;
#pragma unroll
  for (int j = 0; j < 4; j++) Br[j] = Bb + (size_t)(bn + j*16 + lh) * ldb;

  v8f zero = {};
  v8f acc[4][4];
#pragma unroll
  for (int i = 0; i < 4; i++)
#pragma unroll
    for (int j = 0; j < 4; j++) acc[i][j] = zero;

  for (int k0 = 0; k0 < K; k0 += 32) {
    HV16 af[4], bf[4];
#pragma unroll
    for (int i = 0; i < 4; i++) {
      // 16-bit A fragment: lanes 0-15 hold K[k0..k0+7],[k0+16..k0+23];
      // lanes 16-31 hold K[k0+8..k0+15],[k0+24..k0+31]
      af[i].h[0] = *(const v8h*)(Ar[i] + k0 + hi*8);
      af[i].h[1] = *(const v8h*)(Ar[i] + k0 + 16 + hi*8);
    }
#pragma unroll
    for (int j = 0; j < 4; j++) {
      // 16-bit B fragment: same K order as A (K[k0+hi*8..+7],[k0+16+hi*8..+7])
      bf[j].h[0] = *(const v8h*)(Br[j] + k0 + hi*8);
      bf[j].h[1] = *(const v8h*)(Br[j] + k0 + 16 + hi*8);
    }
#pragma unroll
    for (int i = 0; i < 4; i++)
#pragma unroll
      for (int j = 0; j < 4; j++)
        acc[i][j] = __builtin_amdgcn_wmma_f32_16x16x32_f16(
            false, af[i].v, false, bf[j].v, (short)0, acc[i][j], false, false);
  }

  _Float16* Cb = nullptr;
  if (EPI != 4) Cb = Co + (ull)z * sC;

#pragma unroll
  for (int i = 0; i < 4; i++) {
#pragma unroll
    for (int r = 0; r < 8; r++) {
      const int m = bm + i*16 + hi*8 + r;    // C/D layout: M = r + hi*8
      float gmv = 1.f;
      if (EPI == 3) gmv = gate[m];
#pragma unroll
      for (int j = 0; j < 4; j++) {
        const int n = bn + j*16 + lh;        // C/D layout: N = lane%16
        float x = acc[i][j][r];
        if (EPI == 0) {
          if (bias) x += bias[n];
          Cb[(size_t)m * ldc + n] = (_Float16)x;
        } else if (EPI == 1) {
          x = seluf(x + bias[n]);
          Cb[(size_t)m * ldc + n] = (_Float16)x;
        } else if (EPI == 2) {
          Cb[(size_t)m * ldc + n] = (_Float16)(x * alpha);
        } else if (EPI == 3) {
          x = gamma[n] * BN_INV * (gmv * x + bias[n]) + beta[n];
          x = fmaxf(x, 0.f);
          Cb[(size_t)m * ldc + n] = (_Float16)x;
        } else { // EPI == 4 : head, f32 transposed store to [B, N, P]
          x = gamma[n] * BN_INV * (x + bias[n]) + beta[n];
          x = fmaxf(x, 0.f);
          const int bb = m >> 10;            // P = 1024
          const int p  = m & 1023;
          Cf[((size_t)(bb * ldc + n) << 10) | (unsigned)p] = x;
        }
      }
    }
  }
}

// ---------------------------------------------------------------------------
// Helper kernels
// ---------------------------------------------------------------------------
__global__ void f32_to_f16(const float* __restrict__ in, _Float16* __restrict__ out, size_t n){
  size_t i = (size_t)blockIdx.x * blockDim.x + threadIdx.x;
  size_t st = (size_t)gridDim.x * blockDim.x;
  for (; i < n; i += st) out[i] = (_Float16)in[i];
}

// f32 [B,C,P] -> f16 [B,P,C] (channels-last), 32x32 LDS tiles
__global__ void transpose_cl(const float* __restrict__ in, _Float16* __restrict__ out,
                             int C, int P){
  __shared__ _Float16 t[32][33];
  const int b = blockIdx.z;
  const int p0 = blockIdx.x * 32, c0 = blockIdx.y * 32;
  const float* ib = in + (size_t)b * C * P;
  _Float16* ob = out + (size_t)b * P * C;
#pragma unroll
  for (int r = 0; r < 32; r += 8)
    t[threadIdx.y + r][threadIdx.x] =
        (_Float16)ib[(size_t)(c0 + threadIdx.y + r) * P + p0 + threadIdx.x];
  __syncthreads();
#pragma unroll
  for (int r = 0; r < 32; r += 8)
    ob[(size_t)(p0 + threadIdx.y + r) * C + c0 + threadIdx.x] = t[threadIdx.x][threadIdx.y + r];
}

// softmax over last dim, rows of exactly 256 halves; 1 wave / row, 4 rows / block
__global__ __launch_bounds__(128)
void softmax_c256(const _Float16* __restrict__ x, _Float16* __restrict__ out){
  const int row  = blockIdx.x * 4 + (threadIdx.x >> 5);
  const int lane = threadIdx.x & 31;
  const _Float16* xr = x + (size_t)row * 256 + lane * 8;
  float v[8]; float m = -3.0e38f;
#pragma unroll
  for (int i = 0; i < 8; i++){ v[i] = (float)xr[i]; m = fmaxf(m, v[i]); }
#pragma unroll
  for (int o = 16; o > 0; o >>= 1) m = fmaxf(m, __shfl_xor(m, o, 32));
  float s = 0.f;
#pragma unroll
  for (int i = 0; i < 8; i++){ v[i] = __expf(v[i] - m); s += v[i]; }
#pragma unroll
  for (int o = 16; o > 0; o >>= 1) s += __shfl_xor(s, o, 32);
  const float inv = 1.f / s;
  _Float16* orow = out + (size_t)row * 256 + lane * 8;
#pragma unroll
  for (int i = 0; i < 8; i++) orow[i] = (_Float16)(v[i] * inv);
}

// softmax over P axis of channels-last [B,P,C]; block (32c x 8p), grid (C/32, B)
__global__ void softmax_p(const _Float16* __restrict__ x, _Float16* __restrict__ out,
                          int P, int C){
  const int b  = blockIdx.y;
  const int c  = blockIdx.x * 32 + threadIdx.x;
  const int tp = threadIdx.y;
  const size_t base = (size_t)b * P * C + c;
  __shared__ float red[8][32];
  float m = -3.0e38f;
  for (int p = tp; p < P; p += 8) m = fmaxf(m, (float)x[base + (size_t)p * C]);
  red[tp][threadIdx.x] = m; __syncthreads();
  if (tp == 0){ float mm = red[0][threadIdx.x];
    for (int i = 1; i < 8; i++) mm = fmaxf(mm, red[i][threadIdx.x]);
    red[0][threadIdx.x] = mm; }
  __syncthreads();
  m = red[0][threadIdx.x]; __syncthreads();
  float s = 0.f;
  for (int p = tp; p < P; p += 8){
    float e = __expf((float)x[base + (size_t)p * C] - m);
    out[base + (size_t)p * C] = (_Float16)e;
    s += e;
  }
  red[tp][threadIdx.x] = s; __syncthreads();
  if (tp == 0){ float ss = 0.f;
    for (int i = 0; i < 8; i++) ss += red[i][threadIdx.x];
    red[0][threadIdx.x] = ss; }
  __syncthreads();
  const float inv = 1.f / red[0][threadIdx.x];
  for (int p = tp; p < P; p += 8)
    out[base + (size_t)p * C] = (_Float16)((float)out[base + (size_t)p * C] * inv);
}

// out = x * 0.5/max(||x||_p, 1e-12) + base   (norm over P per (b,c))
__global__ void l2norm_res(const _Float16* __restrict__ x, const _Float16* __restrict__ base,
                           _Float16* __restrict__ out, int P, int C){
  const int b  = blockIdx.y;
  const int c  = blockIdx.x * 32 + threadIdx.x;
  const int tp = threadIdx.y;
  const size_t base_off = (size_t)b * P * C + c;
  __shared__ float red[8][32];
  float s = 0.f;
  for (int p = tp; p < P; p += 8){ float v = (float)x[base_off + (size_t)p * C]; s += v * v; }
  red[tp][threadIdx.x] = s; __syncthreads();
  if (tp == 0){ float t = 0.f;
    for (int i = 0; i < 8; i++) t += red[i][threadIdx.x];
    red[0][threadIdx.x] = t; }
  __syncthreads();
  const float nrm = sqrtf(red[0][threadIdx.x]);
  const float inv = 0.5f / fmaxf(nrm, 1e-12f);
  for (int p = tp; p < P; p += 8){
    const size_t o = base_off + (size_t)p * C;
    out[o] = (_Float16)((float)x[o] * inv + (float)base[o]);
  }
}

// gate = sigmoid(fc5(fc4(f[:,coff:coff+128]))) ; 1 wave / row
__global__ __launch_bounds__(128)
void gate_kernel(const _Float16* __restrict__ f, int ldf, int coff,
                 const float* __restrict__ W4, const float* __restrict__ b4,
                 const float* __restrict__ W5, const float* __restrict__ b5,
                 float* __restrict__ g){
  const int row  = blockIdx.x * 4 + (threadIdx.x >> 5);
  const int lane = threadIdx.x & 31;
  const _Float16* fr = f + (size_t)row * ldf + coff;
  float h0 = b4[lane], h1 = b4[lane + 32];
  for (int k = 0; k < 128; k++){
    const float xv = (float)fr[k];
    h0 += W4[lane * 128 + k] * xv;
    h1 += W4[(lane + 32) * 128 + k] * xv;
  }
  float part = h0 * W5[lane] + h1 * W5[lane + 32];
#pragma unroll
  for (int o = 16; o > 0; o >>= 1) part += __shfl_xor(part, o, 32);
  if (lane == 0) g[row] = 1.f / (1.f + __expf(-(part + b5[0])));
}

// dst[m, 0:Ch]=a[m,:], dst[m, Ch:2Ch]=b[m,:]
__global__ void concat2(const _Float16* __restrict__ a, const _Float16* __restrict__ b,
                        _Float16* __restrict__ dst, size_t M, int Ch){
  const size_t total = M * (size_t)(2 * Ch);
  size_t i = (size_t)blockIdx.x * blockDim.x + threadIdx.x;
  const size_t st = (size_t)gridDim.x * blockDim.x;
  for (; i < total; i += st){
    const size_t m = i / (size_t)(2 * Ch);
    const int c = (int)(i - m * (size_t)(2 * Ch));
    dst[i] = (c < Ch) ? a[m * (size_t)Ch + c] : b[m * (size_t)Ch + (c - Ch)];
  }
}

// ---------------------------------------------------------------------------
static void gemmL(int epi,
                  const _Float16* A, ull sA, int lda,
                  const _Float16* W, ull sB, int ldb,
                  _Float16* Co, ull sC, int ldc, float* Cf,
                  int Mr, int N, int K, int batches,
                  const float* bias, const float* gamma, const float* beta,
                  const float* gate, float alpha, hipStream_t s)
{
  dim3 grid(Mr / 128, N / 128, batches), blk(128);
  switch (epi) {
    case 0: gemm_wmma<0><<<grid, blk, 0, s>>>(A,sA,lda,W,sB,ldb,Co,sC,ldc,Cf,K,bias,gamma,beta,gate,alpha); break;
    case 1: gemm_wmma<1><<<grid, blk, 0, s>>>(A,sA,lda,W,sB,ldb,Co,sC,ldc,Cf,K,bias,gamma,beta,gate,alpha); break;
    case 2: gemm_wmma<2><<<grid, blk, 0, s>>>(A,sA,lda,W,sB,ldb,Co,sC,ldc,Cf,K,bias,gamma,beta,gate,alpha); break;
    case 3: gemm_wmma<3><<<grid, blk, 0, s>>>(A,sA,lda,W,sB,ldb,Co,sC,ldc,Cf,K,bias,gamma,beta,gate,alpha); break;
    case 4: gemm_wmma<4><<<grid, blk, 0, s>>>(A,sA,lda,W,sB,ldb,Co,sC,ldc,Cf,K,bias,gamma,beta,gate,alpha); break;
  }
}

extern "C" void kernel_launch(void* const* d_in, const int* in_sizes, int n_in,
                              void* d_out, int out_size, void* d_ws, size_t ws_size,
                              hipStream_t stream)
{
  (void)in_sizes; (void)n_in; (void)out_size; (void)ws_size;

  // ---- input mapping (setup_inputs() insertion order, recursively flattened)
  const float* pse_in = (const float*)d_in[0];
  const float* val_in = (const float*)d_in[1];
  int ix = 2;
  const float *tW[7], *tb[7];
  for (int i = 0; i < 7; i++){ tW[i] = (const float*)d_in[ix++]; tb[i] = (const float*)d_in[ix++]; }
  struct RoiP {
    const float *fc1W,*fc1b,*fc2W,*fc2b,*fc3W,*fc3b;
    const float *fc4pW,*fc4pb,*fc4vW,*fc4vb,*fc5pW,*fc5pb,*fc5vW,*fc5vb;
    const float *c1W,*c1b,*bn1g,*bn1b,*c2W,*c2b,*bn2g,*bn2b;
  } roi[2];
  for (int r = 0; r < 2; r++){
    RoiP& q = roi[r];
    q.fc1W=(const float*)d_in[ix++]; q.fc1b=(const float*)d_in[ix++];
    q.fc2W=(const float*)d_in[ix++]; q.fc2b=(const float*)d_in[ix++];
    q.fc3W=(const float*)d_in[ix++]; q.fc3b=(const float*)d_in[ix++];
    q.fc4pW=(const float*)d_in[ix++]; q.fc4pb=(const float*)d_in[ix++];
    q.fc4vW=(const float*)d_in[ix++]; q.fc4vb=(const float*)d_in[ix++];
    q.fc5pW=(const float*)d_in[ix++]; q.fc5pb=(const float*)d_in[ix++];
    q.fc5vW=(const float*)d_in[ix++]; q.fc5vb=(const float*)d_in[ix++];
    q.c1W=(const float*)d_in[ix++];  q.c1b=(const float*)d_in[ix++];
    q.bn1g=(const float*)d_in[ix++]; q.bn1b=(const float*)d_in[ix++];
    q.c2W=(const float*)d_in[ix++];  q.c2b=(const float*)d_in[ix++];
    q.bn2g=(const float*)d_in[ix++]; q.bn2b=(const float*)d_in[ix++];
  }
  const float* hW    = (const float*)d_in[ix++];
  const float* hb    = (const float*)d_in[ix++];
  const float* hg    = (const float*)d_in[ix++];
  const float* hbeta = (const float*)d_in[ix++];

  // ---- workspace bump allocator (peak ~245 MB)
  char* wsp = (char*)d_ws;
  size_t off = 0;
  auto allocB = [&](size_t bytes)->void* {
    void* p = wsp + off; off += (bytes + 255) & ~(size_t)255; return p;
  };
  auto allocH = [&](size_t nh)->_Float16* { return (_Float16*)allocB(nh * 2); };
  auto allocF = [&](size_t nf)->float*    { return (float*)allocB(nf * 4); };

  auto cvt = [&](const float* src, size_t n)->_Float16* {
    _Float16* dst = allocH(n);
    f32_to_f16<<<dim3(512), dim3(256), 0, stream>>>(src, dst, n);
    return dst;
  };

  // f16 weights (K-contiguous [N,K] row-major, same as torch layout)
  _Float16* tW16[7];
  for (int i = 0; i < 7; i++) tW16[i] = cvt(tW[i], (size_t)KC * KC);
  _Float16* roiW16[2][5];
  for (int r = 0; r < 2; r++){
    roiW16[r][0] = cvt(roi[r].fc1W, (size_t)4*KC * 2*KC);
    roiW16[r][1] = cvt(roi[r].fc2W, (size_t)2*KC * 4*KC);
    roiW16[r][2] = cvt(roi[r].fc3W, (size_t)KC * 2*KC);
    roiW16[r][3] = cvt(roi[r].c1W,  (size_t)KC * KC);
    roiW16[r][4] = cvt(roi[r].c2W,  (size_t)KC * KC);
  }
  _Float16* hW16 = cvt(hW, (size_t)KOUT * 2*KC);

  const size_t MC = (size_t)KM * KC;       // 8,388,608 halves per [M,C] buffer
  _Float16* bb = allocH(8 * MC);           // 8 contiguous [M,C] buffers
  _Float16* Bf[8];
  for (int i = 0; i < 8; i++) Bf[i] = bb + (size_t)i * MC;
  _Float16* tA   = allocH(MC);
  _Float16* tB   = allocH(MC);
  _Float16* Abig = allocH((size_t)KB * KP * KP);  // attention matrix / concat scratch
  float* gp = allocF(KM);
  float* gv = allocF(KM);

  // ---- inputs: channels-last f16 (b0,b1) + channels-first f16 (b2,b3)
  transpose_cl<<<dim3(KP/32, KC/32, KB), dim3(32, 8), 0, stream>>>(pse_in, Bf[0], KC, KP);
  transpose_cl<<<dim3(KP/32, KC/32, KB), dim3(32, 8), 0, stream>>>(val_in, Bf[1], KC, KP);
  f32_to_f16<<<dim3(1024), dim3(256), 0, stream>>>(pse_in, Bf[2], MC);
  f32_to_f16<<<dim3(1024), dim3(256), 0, stream>>>(val_in, Bf[3], MC);

  // shared trans_fc1 stack: 4 linears, SELU folded into 4th, 3 linears
  auto chain = [&](const _Float16* x0, _Float16* u, _Float16* v)->_Float16* {
    const _Float16* x = x0;
    for (int s = 0; s < 7; s++){
      _Float16* o = (s & 1) ? v : u;
      gemmL(s == 3 ? 1 : 0, x, 0, KC, tW16[s], 0, KC, o, 0, KC, nullptr,
            KM, KC, KC, 1, tb[s], nullptr, nullptr, nullptr, 1.f, stream);
      x = o;
    }
    return (_Float16*)x;   // ends in u
  };

  // ===== trans_attention =====
  _Float16* fp = chain(Bf[0], tA, tB);                                  // fp in tA
  softmax_p  <<<dim3(KC/32, KB), dim3(32, 8), 0, stream>>>(fp, Bf[4], KP, KC); // Qp
  softmax_c256<<<dim3(KM/4), dim3(128), 0, stream>>>(fp, Bf[5]);               // Kp
  _Float16* fv = chain(Bf[1], tA, tB);                                  // fv in tA
  softmax_p  <<<dim3(KC/32, KB), dim3(32, 8), 0, stream>>>(fv, Bf[6], KP, KC); // Qv
  softmax_c256<<<dim3(KM/4), dim3(128), 0, stream>>>(fv, Bf[7]);               // Kv

  // pseudo branch: Ap = Qp@Kv^T ; op = (Ap@xp)/dn ; trans_fc1 ; l2norm*0.5+pse0
  gemmL(0, Bf[4], (ull)KP*KC, KC, Bf[7], (ull)KP*KC, KC,
        Abig, (ull)KP*KP, KP, nullptr, KP, KP, KC, KB,
        nullptr, nullptr, nullptr, nullptr, 1.f, stream);
  gemmL(2, Abig, (ull)KP*KP, KP, Bf[2], (ull)KC*KP, KP,
        tA, (ull)KP*KC, KC, nullptr, KP, KC, KP, KB,
        nullptr, nullptr, nullptr, nullptr, 1.f/256.f, stream);
  _Float16* opf = chain(tA, tB, tA);                                    // in tB
  l2norm_res<<<dim3(KC/32, KB), dim3(32, 8), 0, stream>>>(opf, Bf[0], Bf[4], KP, KC); // new pse -> b4

  // valid branch: Av = Qv@Kp^T ; ov = (Av@xv)/dn ; trans_fc1 ; l2norm*0.5+val0
  gemmL(0, Bf[6], (ull)KP*KC, KC, Bf[5], (ull)KP*KC, KC,
        Abig, (ull)KP*KP, KP, nullptr, KP, KP, KC, KB,
        nullptr, nullptr, nullptr, nullptr, 1.f, stream);
  gemmL(2, Abig, (ull)KP*KP, KP, Bf[3], (ull)KC*KP, KP,
        tA, (ull)KP*KC, KC, nullptr, KP, KC, KP, KB,
        nullptr, nullptr, nullptr, nullptr, 1.f/256.f, stream);
  _Float16* ovf = chain(tA, tB, tA);                                    // in tB
  l2norm_res<<<dim3(KC/32, KB), dim3(32, 8), 0, stream>>>(ovf, Bf[1], Bf[5], KP, KC); // new val -> b5

  // ===== roi_attention (x2) =====
  auto run_roi = [&](const RoiP& q, _Float16* const w16[5],
                     const _Float16* pse, const _Float16* val,
                     _Float16* t4, _Float16* t2, _Float16* outP, _Float16* outV){
    _Float16* fcat = Abig;
    concat2<<<dim3(4096), dim3(256), 0, stream>>>(pse, val, fcat, (size_t)KM, KC);
    gemmL(0, fcat, 0, 2*KC, w16[0], 0, 2*KC, t4, 0, 4*KC, nullptr,
          KM, 4*KC, 2*KC, 1, q.fc1b, nullptr, nullptr, nullptr, 1.f, stream);
    gemmL(0, t4, 0, 4*KC, w16[1], 0, 4*KC, t2, 0, 2*KC, nullptr,
          KM, 2*KC, 4*KC, 1, q.fc2b, nullptr, nullptr, nullptr, 1.f, stream);
    gemmL(0, t2, 0, 2*KC, w16[2], 0, 2*KC, tA, 0, KC, nullptr,
          KM, KC, 2*KC, 1, q.fc3b, nullptr, nullptr, nullptr, 1.f, stream);
    gate_kernel<<<dim3(KM/4), dim3(128), 0, stream>>>(tA, KC, 0,
        q.fc4pW, q.fc4pb, q.fc5pW, q.fc5pb, gp);
    gate_kernel<<<dim3(KM/4), dim3(128), 0, stream>>>(tA, KC, 128,
        q.fc4vW, q.fc4vb, q.fc5vW, q.fc5vb, gv);
    gemmL(3, pse, 0, KC, w16[3], 0, KC, outP, 0, KC, nullptr,
          KM, KC, KC, 1, q.c1b, q.bn1g, q.bn1b, gp, 1.f, stream);
    gemmL(3, val, 0, KC, w16[4], 0, KC, outV, 0, KC, nullptr,
          KM, KC, KC, 1, q.c2b, q.bn2g, q.bn2b, gv, 1.f, stream);
  };

  // roi1: pse=b4 val=b5 ; t4 spans b0..b3, t2 spans b6..b7 ; out -> b0,b1
  run_roi(roi[0], roiW16[0], Bf[4], Bf[5], Bf[0], Bf[6], Bf[0], Bf[1]);
  // roi2: pse=b0 val=b1 ; t4 spans b4..b7, t2 spans b2..b3 ; out -> b2,b3
  run_roi(roi[1], roiW16[1], Bf[0], Bf[1], Bf[4], Bf[2], Bf[2], Bf[3]);

  // ===== head: fusion = concat([val, pse], channels) ; conv+BN+ReLU -> [B,OUT,P] f32
  concat2<<<dim3(4096), dim3(256), 0, stream>>>(Bf[3], Bf[2], Abig, (size_t)KM, KC);
  gemmL(4, Abig, 0, 2*KC, hW16, 0, 2*KC, nullptr, 0, KOUT, (float*)d_out,
        KM, KOUT, 2*KC, 1, hb, hg, hbeta, nullptr, 1.f, stream);
}
